// QWenAttention_13529146982628
// MI455X (gfx1250) — compile-verified, matrix operand order checked by reading, not run on hardware
//
#include <hip/hip_runtime.h>
#include <hip/hip_bf16.h>

typedef __attribute__((ext_vector_type(16))) __bf16 v16bf;
typedef __attribute__((ext_vector_type(8)))  __bf16 v8bf;
typedef __attribute__((ext_vector_type(8)))  float  v8f;
typedef __attribute__((ext_vector_type(4)))  int    v4i;

constexpr int BB   = 2;
constexpr int SS   = 2048;
constexpr int HIDC = 2048;
constexpr int NHC  = 16;
constexpr int HDC  = 128;
constexpr int MROWS = BB * SS;                 // 4096 token rows
constexpr float SCALE = 0.08838834764831845f;  // 1/sqrt(128)

#if defined(__has_builtin)
#if __has_builtin(__builtin_amdgcn_global_load_async_to_lds_b128)
#define HAVE_ASYNC_LDS 1
#endif
#endif

// Assemble a 16x32 bf16 WMMA operand fragment for this lane.
// ISA layout (CDNA5 ISA): lanes 0-15 hold K {0..7,16..23}, lanes 16-31 hold
// K {8..15,24..31}; caller passes p already offset by (lane<16?0:8).
__device__ __forceinline__ v16bf pack16(const __bf16* p) {
  v8bf lo = *(const v8bf*)(p);
  v8bf hi = *(const v8bf*)(p + 16);
  v16bf r;
#pragma unroll
  for (int j = 0; j < 8; ++j) { r[j] = lo[j]; r[j + 8] = hi[j]; }
  return r;
}

// ---------------------------------------------------------------------------
// Tiled GEMM: C[M,N] = A[M,K] @ W[K,N] (+ bias).  A is float or bf16 row-major,
// W is fp32 row-major, C is fp32 or bf16.  Block tile 128x128, K-step 32,
// double-buffered LDS (stage tile k+1 while WMMAs consume tile k),
// 8 waves each owning a 32x64 sub-tile (2x4 grid of 16x16 WMMA accumulators).
// ---------------------------------------------------------------------------
template <typename AT, bool CBF16, bool BIAS>
__global__ __launch_bounds__(256)
void gemm_kernel(const AT* __restrict__ A, const float* __restrict__ W,
                 const float* __restrict__ bias, void* __restrict__ Cout,
                 int M, int N, int K) {
  __shared__ __bf16 Alds[2][128][32 + 8];   // [buf][m][k]
  __shared__ __bf16 Wlds[2][128][32 + 8];   // [buf][n][k] (W tile transposed)

  const int tid  = threadIdx.x;
  const int lane = tid & 31;
  const int wave = tid >> 5;
  const int wm   = wave >> 1;     // 0..3 -> 32-row band
  const int wn   = wave & 1;      // 0..1 -> 64-col band
  const int l16  = lane & 15;
  const int koff = (lane < 16) ? 0 : 8;
  const int tileM = blockIdx.y * 128;
  const int tileN = blockIdx.x * 128;

  auto stage = [&](int buf, int kt) {
    if constexpr (__is_same(AT, float)) {
#pragma unroll 2
      for (int i = tid; i < 128 * 8; i += 256) {
        const int r = i >> 3, c = (i & 7) * 4;
        const float4 v = *(const float4*)(A + (size_t)(tileM + r) * K + kt + c);
        Alds[buf][r][c + 0] = (__bf16)v.x; Alds[buf][r][c + 1] = (__bf16)v.y;
        Alds[buf][r][c + 2] = (__bf16)v.z; Alds[buf][r][c + 3] = (__bf16)v.w;
      }
    } else {
#pragma unroll 2
      for (int i = tid; i < 128 * 8; i += 256) {
        const int r = i >> 3, c = (i & 7) * 4;
        const AT* p = A + (size_t)(tileM + r) * K + kt + c;
        Alds[buf][r][c + 0] = p[0]; Alds[buf][r][c + 1] = p[1];
        Alds[buf][r][c + 2] = p[2]; Alds[buf][r][c + 3] = p[3];
      }
    }
#pragma unroll 2
    for (int i = tid; i < 32 * 32; i += 256) {
      const int kr = i >> 5, c = (i & 31) * 4;
      const float4 v = *(const float4*)(W + (size_t)(kt + kr) * N + tileN + c);
      Wlds[buf][c + 0][kr] = (__bf16)v.x; Wlds[buf][c + 1][kr] = (__bf16)v.y;
      Wlds[buf][c + 2][kr] = (__bf16)v.z; Wlds[buf][c + 3][kr] = (__bf16)v.w;
    }
  };

  v8f acc[2][4];
#pragma unroll
  for (int mi = 0; mi < 2; ++mi)
#pragma unroll
    for (int ni = 0; ni < 4; ++ni) acc[mi][ni] = (v8f)0.0f;

  stage(0, 0);
  __syncthreads();

  int buf = 0;
  for (int kt = 0; kt < K; kt += 32) {
    if (kt + 32 < K) stage(buf ^ 1, kt + 32);  // prefetch next tile

    v16bf am[2], bn[4];
#pragma unroll
    for (int mi = 0; mi < 2; ++mi)
      am[mi] = pack16(&Alds[buf][wm * 32 + mi * 16 + l16][koff]);
#pragma unroll
    for (int ni = 0; ni < 4; ++ni)
      bn[ni] = pack16(&Wlds[buf][wn * 64 + ni * 16 + l16][koff]);

#pragma unroll
    for (int mi = 0; mi < 2; ++mi)
#pragma unroll
      for (int ni = 0; ni < 4; ++ni)
        acc[mi][ni] = __builtin_amdgcn_wmma_f32_16x16x32_bf16(
            false, am[mi], false, bn[ni], (short)0, acc[mi][ni], false, false);

    __syncthreads();
    buf ^= 1;
  }

  // Epilogue: C/D layout = VGPR r -> row r (lanes 0-15) / r+8 (lanes 16-31).
#pragma unroll
  for (int mi = 0; mi < 2; ++mi)
#pragma unroll
    for (int ni = 0; ni < 4; ++ni) {
      const int col = tileN + wn * 64 + ni * 16 + l16;
      const float bv = BIAS ? bias[col] : 0.0f;
#pragma unroll
      for (int r = 0; r < 8; ++r) {
        const int row = tileM + wm * 32 + mi * 16 + r + ((lane < 16) ? 0 : 8);
        const float val = acc[mi][ni][r] + bv;
        if constexpr (CBF16)
          ((__bf16*)Cout)[(size_t)row * N + col] = (__bf16)val;
        else
          ((float*)Cout)[(size_t)row * N + col] = val;
      }
    }
}

// ---------------------------------------------------------------------------
// NeoX RoPE applied in place to bf16 Q and K.  One thread per (token,head,i<64)
// pair; each thread owns both halves of its rotation pair, so in-place is safe.
// ---------------------------------------------------------------------------
__global__ __launch_bounds__(256)
void rope_kernel(__bf16* __restrict__ Q, __bf16* __restrict__ K,
                 const int* __restrict__ positions) {
  const int idx = blockIdx.x * 256 + threadIdx.x;  // over MROWS*NHC*64
  const int i  = idx & 63;
  const int h  = (idx >> 6) & (NHC - 1);
  const int bs = idx >> 10;                        // 0..MROWS-1
  const float p = (float)positions[bs];
  const float inv = __powf(10000.0f, -(float)i * (1.0f / 64.0f));
  const float f = p * inv;
  const float c = __cosf(f), s = __sinf(f);
  const size_t base = (size_t)bs * HIDC + h * HDC;

  const float q1 = (float)Q[base + i], q2 = (float)Q[base + i + 64];
  Q[base + i]      = (__bf16)(q1 * c - q2 * s);
  Q[base + i + 64] = (__bf16)(q2 * c + q1 * s);
  const float k1 = (float)K[base + i], k2 = (float)K[base + i + 64];
  K[base + i]      = (__bf16)(k1 * c - k2 * s);
  K[base + i + 64] = (__bf16)(k2 * c + k1 * s);
}

// ---------------------------------------------------------------------------
// Causal flash attention.  One block per (b, h, 128 query rows); each of the
// 8 waves owns 16 query rows.  Loop over 32-key blocks staged in LDS.
// K staging uses GLOBAL_LOAD_ASYNC_TO_LDS (ASYNCcnt) when available.
// ---------------------------------------------------------------------------
__global__ __launch_bounds__(256)
void attn_kernel(const __bf16* __restrict__ Q, const __bf16* __restrict__ Kb,
                 const __bf16* __restrict__ Vb, __bf16* __restrict__ O) {
  __shared__ __bf16 Klds[32][HDC + 8];       // [key][d]  (N-major for Q@K^T)
  __shared__ __bf16 Vlds[HDC][32 + 8];       // [d][key]  (N-major for P@V)
  __shared__ __bf16 Plds[8][16][32 + 8];     // per-wave P tile round-trip

  const int tid  = threadIdx.x;
  const int lane = tid & 31;
  const int wave = tid >> 5;
  const int l16  = lane & 15;
  const int koff = (lane < 16) ? 0 : 8;
  const int rofs = (lane < 16) ? 0 : 8;
  const int qb = blockIdx.x * 128;
  const int h  = blockIdx.y;
  const int b  = blockIdx.z;

  // Q fragments (16 rows x 128 dims) straight from global: row-major layout
  // matches the A-fragment pattern.
  v16bf aq[4];
  const __bf16* qptr =
      Q + ((size_t)(b * SS + qb + wave * 16 + l16) * HIDC + h * HDC);
#pragma unroll
  for (int kc = 0; kc < 4; ++kc) aq[kc] = pack16(qptr + kc * 32 + koff);

  v8f o[8];
#pragma unroll
  for (int n = 0; n < 8; ++n) o[n] = (v8f)0.0f;
  float m_s[8], l_s[8];
#pragma unroll
  for (int r = 0; r < 8; ++r) { m_s[r] = -1e30f; l_s[r] = 0.0f; }

  const int nkb = (qb >> 5) + 4;  // key blocks covering [0, qb+128)
  for (int kb = 0; kb < nkb; ++kb) {
    const int k0 = kb * 32;
    // Stage K block row-major (raw bf16 copy -> async LDS DMA when available).
#ifdef HAVE_ASYNC_LDS
#pragma unroll 2
    for (int i = tid; i < 32 * 16; i += 256) {
      const int r = i >> 4, c = (i & 15) * 8;
      const __bf16* g =
          Kb + ((size_t)(b * SS + k0 + r) * HIDC + h * HDC + c);
      __builtin_amdgcn_global_load_async_to_lds_b128(
          (__attribute__((address_space(1))) v4i*)g,
          (__attribute__((address_space(3))) v4i*)&Klds[r][c], 0, 0);
    }
#else
#pragma unroll 2
    for (int i = tid; i < 32 * 16; i += 256) {
      const int r = i >> 4, c = (i & 15) * 8;
      *(v8bf*)&Klds[r][c] =
          *(const v8bf*)(Kb + ((size_t)(b * SS + k0 + r) * HIDC + h * HDC + c));
    }
#endif
    // Stage V block transposed.
#pragma unroll 4
    for (int i = tid; i < 32 * HDC; i += 256) {
      const int key = i & 31, d = i >> 5;
      Vlds[d][key] = Vb[(size_t)(b * SS + k0 + key) * HIDC + h * HDC + d];
    }
#ifdef HAVE_ASYNC_LDS
    asm volatile("s_wait_asynccnt 0" ::: "memory");
#endif
    __syncthreads();

    // S = Q @ K^T  (16 rows x 32 keys, K-dim = 128).  Batch the B-fragment
    // loads so ds_load_b128s stay in flight while WMMAs issue.
    v8f sc[2];
    sc[0] = (v8f)0.0f; sc[1] = (v8f)0.0f;
#pragma unroll
    for (int n = 0; n < 2; ++n) {
      v16bf bk[4];
#pragma unroll
      for (int kc = 0; kc < 4; ++kc)
        bk[kc] = pack16(&Klds[n * 16 + l16][kc * 32 + koff]);
#pragma unroll
      for (int kc = 0; kc < 4; ++kc)
        sc[n] = __builtin_amdgcn_wmma_f32_16x16x32_bf16(
            false, aq[kc], false, bk[kc], (short)0, sc[n], false, false);
    }

    // Online softmax per row (rows live in 16-lane halves of the wave).
#pragma unroll
    for (int r = 0; r < 8; ++r) {
      const int qrow = qb + wave * 16 + r + rofs;
      float s0 = sc[0][r] * SCALE;
      if (k0 + l16 > qrow) s0 = -1e30f;
      float s1 = sc[1][r] * SCALE;
      if (k0 + 16 + l16 > qrow) s1 = -1e30f;
      float mx = fmaxf(s0, s1);
      mx = fmaxf(mx, __shfl_xor(mx, 1, 16));
      mx = fmaxf(mx, __shfl_xor(mx, 2, 16));
      mx = fmaxf(mx, __shfl_xor(mx, 4, 16));
      mx = fmaxf(mx, __shfl_xor(mx, 8, 16));
      const float mnew = fmaxf(m_s[r], mx);
      const float alpha = __expf(m_s[r] - mnew);
      const float p0 = __expf(s0 - mnew);
      const float p1 = __expf(s1 - mnew);
      float ps = p0 + p1;
      ps += __shfl_xor(ps, 1, 16);
      ps += __shfl_xor(ps, 2, 16);
      ps += __shfl_xor(ps, 4, 16);
      ps += __shfl_xor(ps, 8, 16);
      l_s[r] = l_s[r] * alpha + ps;
      m_s[r] = mnew;
#pragma unroll
      for (int n2 = 0; n2 < 8; ++n2) o[n2][r] *= alpha;
      Plds[wave][r + rofs][l16]      = (__bf16)p0;
      Plds[wave][r + rofs][16 + l16] = (__bf16)p1;
    }
    // Within-wave cross-lane LDS dependency: wait our DS stores.
    asm volatile("s_wait_dscnt 0" ::: "memory");

    // O += P @ V  (16 x 128, K-dim = 32); batch fragments in pairs.
    v16bf ap = pack16(&Plds[wave][l16][koff]);
#pragma unroll
    for (int n2 = 0; n2 < 8; n2 += 2) {
      v16bf bv0 = pack16(&Vlds[(n2 + 0) * 16 + l16][koff]);
      v16bf bv1 = pack16(&Vlds[(n2 + 1) * 16 + l16][koff]);
      o[n2 + 0] = __builtin_amdgcn_wmma_f32_16x16x32_bf16(
          false, ap, false, bv0, (short)0, o[n2 + 0], false, false);
      o[n2 + 1] = __builtin_amdgcn_wmma_f32_16x16x32_bf16(
          false, ap, false, bv1, (short)0, o[n2 + 1], false, false);
    }
    __syncthreads();
  }

  // Normalize and write attention output (bf16).
#pragma unroll
  for (int r = 0; r < 8; ++r) {
    const float inv = 1.0f / l_s[r];
    const int qrow = qb + wave * 16 + r + rofs;
#pragma unroll
    for (int n2 = 0; n2 < 8; ++n2)
      O[(size_t)(b * SS + qrow) * HIDC + h * HDC + n2 * 16 + l16] =
          (__bf16)(o[n2][r] * inv);
  }
}

// ---------------------------------------------------------------------------
extern "C" void kernel_launch(void* const* d_in, const int* in_sizes, int n_in,
                              void* d_out, int out_size, void* d_ws,
                              size_t ws_size, hipStream_t stream) {
  const float* hs  = (const float*)d_in[0];
  const int*   pos = (const int*)d_in[1];
  const float* wq  = (const float*)d_in[2];
  const float* bq  = (const float*)d_in[3];
  const float* wk  = (const float*)d_in[4];
  const float* bk  = (const float*)d_in[5];
  const float* wv  = (const float*)d_in[6];
  const float* bv  = (const float*)d_in[7];
  const float* wc  = (const float*)d_in[8];

  const size_t QS = (size_t)MROWS * HIDC;  // elements per activation tensor
  __bf16* Qbf = (__bf16*)d_ws;
  __bf16* Kbf = Qbf + QS;
  __bf16* Vbf = Kbf + QS;
  __bf16* Abf = Vbf + QS;

  const dim3 gGemm(HIDC / 128, MROWS / 128);  // 16 x 32
  gemm_kernel<float, true, true><<<gGemm, 256, 0, stream>>>(
      hs, wq, bq, (void*)Qbf, MROWS, HIDC, HIDC);
  gemm_kernel<float, true, true><<<gGemm, 256, 0, stream>>>(
      hs, wk, bk, (void*)Kbf, MROWS, HIDC, HIDC);
  gemm_kernel<float, true, true><<<gGemm, 256, 0, stream>>>(
      hs, wv, bv, (void*)Vbf, MROWS, HIDC, HIDC);

  rope_kernel<<<(MROWS * NHC * 64) / 256, 256, 0, stream>>>(Qbf, Kbf, pos);

  attn_kernel<<<dim3(SS / 128, NHC, BB), 256, 0, stream>>>(Qbf, Kbf, Vbf, Abf);

  gemm_kernel<__bf16, false, false><<<gGemm, 256, 0, stream>>>(
      Abf, wc, nullptr, d_out, MROWS, HIDC, HIDC);
}
